// DotProductAttention_62448824484126
// MI455X (gfx1250) — matrix operand order repaired and checked by reading, not compiled, not run on hardware
//
#include <hip/hip_runtime.h>

typedef __attribute__((ext_vector_type(16))) __bf16 v16bf;
typedef __attribute__((ext_vector_type(8)))  __bf16 v8bf;
typedef __attribute__((ext_vector_type(4)))  __bf16 v4bf;
typedef __attribute__((ext_vector_type(8)))  float  v8f;
typedef __attribute__((ext_vector_type(4)))  int    v4i;
struct f4 { float x, y, z, w; };

#define SEQ     2048
#define DHEAD   128
#define BATCH   32
#define WAVES   8
#define WAVE_Q  32                    // query rows per wave (2 M-tiles of 16)
#define BLOCK_Q (WAVES * WAVE_Q)      // 256 query rows per workgroup
#define KBLK    32                    // keys per iteration
#define NBLK    (SEQ / KBLK)
#define KB_STRIDE 136                 // bf16 row stride for K tiles (bank spread)
#define VT_STRIDE 40                  // bf16 row stride for V^T tiles
#define P_STRIDE  40                  // bf16 row stride for P tile

#if __has_builtin(__builtin_amdgcn_global_load_async_to_lds_b128) && \
    __has_builtin(__builtin_amdgcn_s_wait_asynccnt)
#define USE_ASYNC 1
typedef __attribute__((address_space(1))) v4i* g_v4i_ptr;
typedef __attribute__((address_space(3))) v4i* l_v4i_ptr;
#else
#define USE_ASYNC 0
#endif

__device__ __forceinline__ v16bf load_v16bf(const __bf16* p) {
    // two aligned 16B LDS loads -> one 16-element bf16 fragment
    v8bf lo = *(const v8bf*)p;
    v8bf hi = *(const v8bf*)(p + 8);
    return __builtin_shufflevector(lo, hi, 0,1,2,3,4,5,6,7,8,9,10,11,12,13,14,15);
}

__global__ __launch_bounds__(256, 1)
void fa_fwd_kernel(const float* __restrict__ Q, const float* __restrict__ K,
                   const float* __restrict__ V, const int* __restrict__ mask,
                   float* __restrict__ out)
{
    __shared__ float  stageK[KBLK][DHEAD];          // 16 KB async DMA landing pad
    __shared__ __bf16 kb16[KBLK][KB_STRIDE];        // K block, row-major bf16
    __shared__ __bf16 vt16[DHEAD][VT_STRIDE];       // V block transposed, bf16
    __shared__ __bf16 ldsP[WAVES][WAVE_Q][P_STRIDE];

    const int tid  = threadIdx.x;
    const int wave = tid >> 5;
    const int lane = tid & 31;
    const int col  = lane & 15;   // column / row-within-16 index
    const int half = lane >> 4;   // 0: lanes 0-15, 1: lanes 16-31

    const int b  = blockIdx.x / (SEQ / BLOCK_Q);
    const int qt = blockIdx.x % (SEQ / BLOCK_Q);
    const int q0 = qt * BLOCK_Q + wave * WAVE_Q;

    const float* Qb = Q + ((size_t)b * SEQ + q0) * DHEAD;
    const float* Kb = K + (size_t)b * SEQ * DHEAD;
    const float* Vb = V + (size_t)b * SEQ * DHEAD;
    const int*   Mb = mask + (size_t)b * SEQ;

    // staging work split (persistent per thread)
    const int krow = tid >> 3;          // K conversion: row in block
    const int kc0  = (tid & 7) * 16;    // K conversion: 16-elem run start
    const int kk0  = (tid >> 5) * 4;    // V prefetch: key offset of 4x4 subtile
    const int d0   = (tid & 31) * 4;    // V prefetch: head-dim offset

    // ---- Q fragments: 2 M-tiles x 4 D-slices, bf16 A-layout (16x32 each).
    // Contiguous k order used here: lanes 0-15 hold K={0..15}, lanes 16-31 K={16..31}.
    v16bf qfrag[2][4];
    {
        const int kb = half * 16;
        #pragma unroll
        for (int mt = 0; mt < 2; ++mt) {
            const float* src = Qb + (mt * 16 + col) * DHEAD;
            #pragma unroll
            for (int s = 0; s < 4; ++s) {
                const f4 a0 = *(const f4*)(src + s * 32 + kb);
                const f4 a1 = *(const f4*)(src + s * 32 + kb + 4);
                const f4 a2 = *(const f4*)(src + s * 32 + kb + 8);
                const f4 a3 = *(const f4*)(src + s * 32 + kb + 12);
                v16bf q;
                q[0]  = (__bf16)a0.x; q[1]  = (__bf16)a0.y;
                q[2]  = (__bf16)a0.z; q[3]  = (__bf16)a0.w;
                q[4]  = (__bf16)a1.x; q[5]  = (__bf16)a1.y;
                q[6]  = (__bf16)a1.z; q[7]  = (__bf16)a1.w;
                q[8]  = (__bf16)a2.x; q[9]  = (__bf16)a2.y;
                q[10] = (__bf16)a2.z; q[11] = (__bf16)a2.w;
                q[12] = (__bf16)a3.x; q[13] = (__bf16)a3.y;
                q[14] = (__bf16)a3.z; q[15] = (__bf16)a3.w;
                qfrag[mt][s] = q;
            }
        }
    }

    // ---- accumulators: per M-tile, 8 C-tiles of 16x16 f32 + softmax stats
    v8f o[2][8];
    float m[2][8], l[2][8];
    #pragma unroll
    for (int mt = 0; mt < 2; ++mt) {
        #pragma unroll
        for (int t = 0; t < 8; ++t) o[mt][t] = (v8f){0.f,0.f,0.f,0.f,0.f,0.f,0.f,0.f};
        #pragma unroll
        for (int r = 0; r < 8; ++r) { m[mt][r] = -1e30f; l[mt][r] = 0.f; }
    }

    // ---- prologue: kick off block 0 transfers (K via async DMA, V into VGPRs)
    #pragma unroll
    for (int i = 0; i < 4; ++i) {
        const int idx = tid + i * 256;
        const int r   = idx >> 5;
        const int c4  = (idx & 31) * 4;
#if USE_ASYNC
        __builtin_amdgcn_global_load_async_to_lds_b128(
            (g_v4i_ptr)(Kb + (size_t)r * DHEAD + c4),
            (l_v4i_ptr)&stageK[r][c4], 0, 0);
#else
        *(f4*)&stageK[r][c4] = *(const f4*)(Kb + (size_t)r * DHEAD + c4);
#endif
    }
    f4 vpre[4];
    #pragma unroll
    for (int i = 0; i < 4; ++i)
        vpre[i] = *(const f4*)(Vb + (size_t)(kk0 + i) * DHEAD + d0);

    for (int jb = 0; jb < NBLK; ++jb) {
        const int j = jb * KBLK;

#if USE_ASYNC
        __builtin_amdgcn_s_wait_asynccnt(0);   // my share of K(jb) DMA done
#endif
        __syncthreads();   // all DMA landed; prev compute done with kb16/vt16

        // ---- conversion phase: stageK -> kb16 (row-major), vpre -> vt16 (transposed)
        #pragma unroll
        for (int h = 0; h < 2; ++h) {
            v8bf w;
            #pragma unroll
            for (int e = 0; e < 8; ++e)
                w[e] = (__bf16)stageK[krow][kc0 + h * 8 + e];
            *(v8bf*)&kb16[krow][kc0 + h * 8] = w;
        }
        {
            float t0[4] = {vpre[0].x, vpre[1].x, vpre[2].x, vpre[3].x};
            float t1[4] = {vpre[0].y, vpre[1].y, vpre[2].y, vpre[3].y};
            float t2[4] = {vpre[0].z, vpre[1].z, vpre[2].z, vpre[3].z};
            float t3[4] = {vpre[0].w, vpre[1].w, vpre[2].w, vpre[3].w};
            v4bf w0, w1, w2, w3;
            #pragma unroll
            for (int i = 0; i < 4; ++i) {
                w0[i] = (__bf16)t0[i]; w1[i] = (__bf16)t1[i];
                w2[i] = (__bf16)t2[i]; w3[i] = (__bf16)t3[i];
            }
            *(v4bf*)&vt16[d0 + 0][kk0] = w0;
            *(v4bf*)&vt16[d0 + 1][kk0] = w1;
            *(v4bf*)&vt16[d0 + 2][kk0] = w2;
            *(v4bf*)&vt16[d0 + 3][kk0] = w3;
        }
        __syncthreads();   // bf16 tiles ready; stageK free for next DMA

        // ---- kick off NEXT block's transfers; DMA + VMEM overlap the compute below
        if (jb + 1 < NBLK) {
            const int jn = j + KBLK;
            #pragma unroll
            for (int i = 0; i < 4; ++i) {
                const int idx = tid + i * 256;
                const int r   = idx >> 5;
                const int c4  = (idx & 31) * 4;
#if USE_ASYNC
                __builtin_amdgcn_global_load_async_to_lds_b128(
                    (g_v4i_ptr)(Kb + (size_t)(jn + r) * DHEAD + c4),
                    (l_v4i_ptr)&stageK[r][c4], 0, 0);
#else
                *(f4*)&stageK[r][c4] = *(const f4*)(Kb + (size_t)(jn + r) * DHEAD + c4);
#endif
            }
            #pragma unroll
            for (int i = 0; i < 4; ++i)
                vpre[i] = *(const f4*)(Vb + (size_t)(jn + kk0 + i) * DHEAD + d0);
        }

        // ---- scores: S[32q x 32k]; each K B-fragment feeds BOTH M-tiles
        v8f s0[2], s1[2];
        #pragma unroll
        for (int mt = 0; mt < 2; ++mt) {
            s0[mt] = (v8f){0.f,0.f,0.f,0.f,0.f,0.f,0.f,0.f};
            s1[mt] = s0[mt];
        }
        #pragma unroll
        for (int s = 0; s < 4; ++s) {
            const v16bf b0 = load_v16bf(&kb16[col][s * 32 + half * 16]);      // keys j..j+15
            const v16bf b1 = load_v16bf(&kb16[16 + col][s * 32 + half * 16]); // keys j+16..j+31
            #pragma unroll
            for (int mt = 0; mt < 2; ++mt) {
                s0[mt] = __builtin_amdgcn_wmma_f32_16x16x32_bf16(false, qfrag[mt][s], false, b0,
                                                                 (short)0, s0[mt], false, false);
                s1[mt] = __builtin_amdgcn_wmma_f32_16x16x32_bf16(false, qfrag[mt][s], false, b1,
                                                                 (short)0, s1[mt], false, false);
            }
        }

        // ---- key-padding mask (column-wise; shared across M-tiles)
        const int mk0 = Mb[j + col];
        const int mk1 = Mb[j + 16 + col];
        const float pf0 = mk0 ? 1.f : 0.f;
        const float pf1 = mk1 ? 1.f : 0.f;

        // ---- online softmax + P write, per M-tile (limits live ranges)
        #pragma unroll
        for (int mt = 0; mt < 2; ++mt) {
            #pragma unroll
            for (int r = 0; r < 8; ++r) {
                s0[mt][r] = mk0 ? s0[mt][r] : -1e30f;
                s1[mt][r] = mk1 ? s1[mt][r] : -1e30f;
            }
            #pragma unroll
            for (int r = 0; r < 8; ++r) {
                float v = fmaxf(s0[mt][r], s1[mt][r]);
                #pragma unroll
                for (int sh = 1; sh < 16; sh <<= 1)
                    v = fmaxf(v, __shfl_xor(v, sh, 16));
                const float mn = fmaxf(m[mt][r], v);
                const float sc = __expf(m[mt][r] - mn);
                m[mt][r] = mn;
                s0[mt][r] = __expf(s0[mt][r] - mn) * pf0;   // hard-zero padded cols
                s1[mt][r] = __expf(s1[mt][r] - mn) * pf1;
                float rs = s0[mt][r] + s1[mt][r];
                #pragma unroll
                for (int sh = 1; sh < 16; sh <<= 1)
                    rs += __shfl_xor(rs, sh, 16);
                l[mt][r] = l[mt][r] * sc + rs;
                #pragma unroll
                for (int t = 0; t < 8; ++t)
                    o[mt][t][r] *= sc;
            }
            // P: C-layout -> bf16 A-layout staging tile
            #pragma unroll
            for (int r = 0; r < 8; ++r) {
                const int row = mt * 16 + r + 8 * half;
                ldsP[wave][row][col]      = (__bf16)s0[mt][r];
                ldsP[wave][row][16 + col] = (__bf16)s1[mt][r];
            }
        }

        // ---- P A-fragments (one per M-tile)
        v16bf pa[2];
        {
            const int kb = half * 16;
            #pragma unroll
            for (int mt = 0; mt < 2; ++mt) {
                const __bf16* pr = &ldsP[wave][mt * 16 + col][0];
                const v8bf lo = *(const v8bf*)(pr + kb);
                const v8bf hi = *(const v8bf*)(pr + kb + 8);
                pa[mt] = __builtin_shufflevector(lo, hi,
                                                 0,1,2,3,4,5,6,7,8,9,10,11,12,13,14,15);
            }
        }

        // ---- O += P(32x32) * V_block(32x128): each V^T fragment feeds both M-tiles
        #pragma unroll
        for (int t = 0; t < 8; ++t) {
            const v16bf vb = load_v16bf(&vt16[t * 16 + col][half * 16]);
            #pragma unroll
            for (int mt = 0; mt < 2; ++mt)
                o[mt][t] = __builtin_amdgcn_wmma_f32_16x16x32_bf16(false, pa[mt], false, vb,
                                                                   (short)0, o[mt][t], false, false);
        }
    }

    // ---- normalize and store
    float* Ob = out + ((size_t)b * SEQ + q0) * DHEAD;
    #pragma unroll
    for (int mt = 0; mt < 2; ++mt) {
        #pragma unroll
        for (int r = 0; r < 8; ++r) {
            const float inv = l[mt][r] > 0.f ? 1.f / l[mt][r] : 0.f;
            const int row = mt * 16 + r + 8 * half;
            #pragma unroll
            for (int t = 0; t < 8; ++t)
                Ob[(size_t)row * DHEAD + t * 16 + col] = o[mt][t][r] * inv;
        }
    }
}

extern "C" void kernel_launch(void* const* d_in, const int* in_sizes, int n_in,
                              void* d_out, int out_size, void* d_ws, size_t ws_size,
                              hipStream_t stream) {
    const float* Q    = (const float*)d_in[0];
    const float* K    = (const float*)d_in[1];
    const float* V    = (const float*)d_in[2];
    const int*   mask = (const int*)d_in[3];
    float* out = (float*)d_out;

    dim3 grid(BATCH * (SEQ / BLOCK_Q));   // 32 * 8 = 256 workgroups
    dim3 block(256);                      // 8 wave32 per WG
    fa_fwd_kernel<<<grid, block, 0, stream>>>(Q, K, V, mask, out);
}
